// CustomConvLayer_43361989820917
// MI455X (gfx1250) — matrix operand order repaired and checked by reading, not compiled, not run on hardware
//
#include <hip/hip_runtime.h>
#include <hip/hip_bf16.h>

#define SEQ   394
#define EMB   300
#define DEPTH 256

typedef __bf16 bf16_t;
typedef __bf16 v16bf __attribute__((ext_vector_type(16)));
typedef __bf16 bf4   __attribute__((ext_vector_type(4)));
typedef float  v8f   __attribute__((ext_vector_type(8)));

typedef unsigned int v4u __attribute__((ext_vector_type(4)));
typedef int          v4i __attribute__((ext_vector_type(4)));
typedef int          v8i __attribute__((ext_vector_type(8)));

struct B32B { uint4 lo, hi; };

__device__ __forceinline__ v16bf lds_ld16(const bf16_t* pLo, const bf16_t* pHi) {
    B32B t;
    t.lo = *reinterpret_cast<const uint4*>(pLo);   // ds_load_b128
    t.hi = *reinterpret_cast<const uint4*>(pHi);   // ds_load_b128
    return __builtin_bit_cast(v16bf, t);
}

// Convert 16 consecutive f32 in LDS to a bf16 WMMA operand (fallback path).
__device__ __forceinline__ v16bf cvt16(const float* p) {
    float4 a = reinterpret_cast<const float4*>(p)[0];
    float4 b = reinterpret_cast<const float4*>(p)[1];
    float4 c = reinterpret_cast<const float4*>(p)[2];
    float4 d = reinterpret_cast<const float4*>(p)[3];
    v16bf t;
    t[0]=(bf16_t)a.x;  t[1]=(bf16_t)a.y;  t[2]=(bf16_t)a.z;  t[3]=(bf16_t)a.w;
    t[4]=(bf16_t)b.x;  t[5]=(bf16_t)b.y;  t[6]=(bf16_t)b.z;  t[7]=(bf16_t)b.w;
    t[8]=(bf16_t)c.x;  t[9]=(bf16_t)c.y;  t[10]=(bf16_t)c.z; t[11]=(bf16_t)c.w;
    t[12]=(bf16_t)d.x; t[13]=(bf16_t)d.y; t[14]=(bf16_t)d.z; t[15]=(bf16_t)d.w;
    return t;
}

// 2-D TDM load: tile (tile0 x tile1) from a (dim0 x dim1) tensor with row
// stride stride0 (all in elements), zero-fill OOB, LDS padding per cfgWord.
__device__ __forceinline__ void tdm_issue(const void* gptr, unsigned ldsOff,
                                          unsigned dim0, unsigned dim1,
                                          unsigned tile0, unsigned tile1,
                                          unsigned stride0, unsigned cfgWord)
{
    unsigned long long ga = (unsigned long long)(uintptr_t)gptr;
    v4u g0;
    g0[0] = 1u;                                   // count=1, user descriptor
    g0[1] = ldsOff;                               // lds_addr (bytes)
    g0[2] = (unsigned)ga;                         // global_addr[31:0]
    g0[3] = (unsigned)((ga >> 32) & 0x01FFFFFFu) | (2u << 30); // addr[56:32]|type=2
    v8i g1;
    g1[0] = (int)cfgWord;                                     // size/pad config
    g1[1] = (int)((dim0 & 0xFFFFu) << 16);                    // tensor_dim0 lo
    g1[2] = (int)((dim0 >> 16) | ((dim1 & 0xFFFFu) << 16));   // dim0 hi | dim1 lo
    g1[3] = (int)((dim1 >> 16) | (tile0 << 16));              // dim1 hi | tile_dim0
    g1[4] = (int)(tile1 & 0xFFFFu);                           // tile_dim1, tile_dim2=0
    g1[5] = (int)stride0;                                     // tensor_dim0_stride
    g1[6] = 0;
    g1[7] = 0;
    v4i z4 = {0, 0, 0, 0};
#if __clang_major__ >= 23
    v8i z8 = {0, 0, 0, 0, 0, 0, 0, 0};
    __builtin_amdgcn_tensor_load_to_lds(g0, g1, z4, z4, z8, 0);
#else
    __builtin_amdgcn_tensor_load_to_lds(g0, g1, z4, z4, 0);
#endif
}

// x: f32 -> bf16 pre-pass (4 elems/thread)
__global__ void __launch_bounds__(256)
cvt_x_bf16(const float* __restrict__ x, bf16_t* __restrict__ o, int n4)
{
    const int i = blockIdx.x * 256 + threadIdx.x;
    if (i < n4) {
        float4 v = reinterpret_cast<const float4*>(x)[i];
        bf4 t;
        t[0] = (bf16_t)v.x; t[1] = (bf16_t)v.y; t[2] = (bf16_t)v.z; t[3] = (bf16_t)v.w;
        reinterpret_cast<bf4*>(o)[i] = t;
    }
}

// One branch of the text-CNN as a Toeplitz GEMM on the WMMA pipe.
//   C[d, i] = sum_k W[d, k] * Xflat[b][i*EMB + k],   k in [0, N*EMB)
//   out[b, outOff + d] = max(0, max_i (C[d,i] + bias[d]))
template<int N, bool XB16>
__launch_bounds__(256, 1)
__global__ void textcnn_wmma(const float* __restrict__ x,
                             const bf16_t* __restrict__ xbf,
                             const float* __restrict__ W,
                             const float* __restrict__ bias,
                             float* __restrict__ out,
                             int outOff)
{
    constexpr int nK     = N * EMB;                 // 900 / 1200 / 1500
    constexpr int Kpad   = ((nK + 31) / 32) * 32;
    constexpr int KP     = Kpad + 8;                // Wlds row pitch (bf16)
    constexpr int KSTEPS = Kpad / 32;
    constexpr int NWIN   = SEQ - N;
    constexpr int NSC    = (NWIN + 255) / 256;      // 256-window superchunks (=2)
    constexpr int XPITCH = XB16 ? 40 : 36;          // LDS row pitch (elements)
    constexpr int XESZ   = XB16 ? 2 : 4;
    constexpr int XBUFB  = 256 * XPITCH * XESZ;     // bytes per X buffer
    // TDM config: data_size | pad_enable | pad_interval | pad_amount
    //  bf16: 2B, pad every 16 DWORDs (64B row) by 4 DWORDs -> 80B pitch
    //  f32 : 4B, pad every 32 DWORDs (128B row) by 4 DWORDs -> 144B pitch
    constexpr unsigned XCFG = XB16
        ? ((1u << 16) | (1u << 20) | (3u << 22) | (3u << 25))
        : ((2u << 16) | (1u << 20) | (4u << 22) | (3u << 25));

    extern __shared__ char smem[];
    bf16_t* Wlds  = reinterpret_cast<bf16_t*>(smem);            // [64][KP] bf16
    char*   Xbase = smem + (size_t)64 * KP * 2;                 // [2][256][XPITCH]
    float*  red   = reinterpret_cast<float*>(Xbase + 2 * XBUFB);// [8][64]

    const int  tid    = threadIdx.x;
    const int  wave   = tid >> 5;
    const int  lane   = tid & 31;
    const int  laneLo = lane & 15;
    const bool hiG    = lane >= 16;

    const int bIdx  = blockIdx.y;
    const int dBase = blockIdx.x * 64;

    const float*  xb32 = x   + (size_t)bIdx * (SEQ * EMB);
    const bf16_t* xb16 = xbf + (size_t)bIdx * (SEQ * EMB);
    const unsigned xfOff = (unsigned)(uintptr_t)(void*)Xbase;

    auto issueX = [&](int sc, int ks, int buf) {
        const void* src = XB16
            ? (const void*)(xb16 + (size_t)sc * 256 * EMB + ks * 32)
            : (const void*)(xb32 + (size_t)sc * 256 * EMB + ks * 32);
        const int remRows = NWIN - sc * 256;
        tdm_issue(src, xfOff + (unsigned)(buf * XBUFB),
                  (unsigned)(nK - ks * 32), (unsigned)remRows,
                  32u, (unsigned)(remRows < 256 ? remRows : 256),
                  (unsigned)EMB, XCFG);
    };

    // kick off the first X tile before staging weights (DMA overlaps)
    if (tid < 32) issueX(0, 0, 0);

    // ---- stage weight M-tile (f32 -> bf16) into LDS, zero-padded in K ----
    for (int idx = tid; idx < 64 * (Kpad / 4); idx += 256) {
        const int k4 = (idx % (Kpad / 4)) * 4;      // nK % 4 == 0
        const int d  = idx / (Kpad / 4);
        float4 v = {0.f, 0.f, 0.f, 0.f};
        if (k4 < nK)
            v = *reinterpret_cast<const float4*>(W + (size_t)(dBase + d) * nK + k4);
        bf4 o;
        o[0] = (bf16_t)v.x; o[1] = (bf16_t)v.y; o[2] = (bf16_t)v.z; o[3] = (bf16_t)v.w;
        *reinterpret_cast<bf4*>(Wlds + d * KP + k4) = o;
    }

    __builtin_amdgcn_s_wait_tensorcnt(0);   // waves 1-7: count already 0
    __syncthreads();                        // Wlds + X buffer 0 ready

    v8f acc0[4], acc1[4], vmax[4];
    #pragma unroll
    for (int mt = 0; mt < 4; ++mt)
        #pragma unroll
        for (int r = 0; r < 8; ++r) vmax[mt][r] = -__builtin_inff();

    int buf = 0;
    for (int sc = 0; sc < NSC; ++sc) {
        #pragma unroll
        for (int mt = 0; mt < 4; ++mt)
            #pragma unroll
            for (int r = 0; r < 8; ++r) { acc0[mt][r] = 0.f; acc1[mt][r] = 0.f; }

        // wave-uniform chunk validity: skip matrix work on fully-padded chunks
        const int  cb   = sc * 256 + wave * 32;
        const bool doC0 = cb < NWIN;
        const bool doC1 = cb + 16 < NWIN;

        for (int ks = 0; ks < KSTEPS; ++ks) {
            const bool more = (ks + 1 < KSTEPS) || (sc + 1 < NSC);
            const int  nsc  = (ks + 1 < KSTEPS) ? sc : sc + 1;
            const int  nks  = (ks + 1 < KSTEPS) ? ks + 1 : 0;
            if (tid < 32 && more) issueX(nsc, nks, buf ^ 1);

            // ---- compute on resident buffer: up to 8 WMMAs, A shared ----
            const int k0 = ks * 32;
            if (doC0 || doC1) {
                v16bf Am[4];
                #pragma unroll
                for (int mt = 0; mt < 4; ++mt) {
                    const bf16_t* arow = Wlds + (mt * 16 + laneLo) * KP
                                       + k0 + (hiG ? 16 : 0);
                    Am[mt] = lds_ld16(arow, arow + 8);
                }
                const char* xrow0 = Xbase + buf * XBUFB
                                  + ((wave * 32 + laneLo) * XPITCH + (hiG ? 16 : 0)) * XESZ;
                if (doC0) {
                    v16bf B0 = XB16
                        ? lds_ld16(reinterpret_cast<const bf16_t*>(xrow0),
                                   reinterpret_cast<const bf16_t*>(xrow0) + 8)
                        : cvt16(reinterpret_cast<const float*>(xrow0));
                    #pragma unroll
                    for (int mt = 0; mt < 4; ++mt)
                        acc0[mt] = __builtin_amdgcn_wmma_f32_16x16x32_bf16(
                            false, Am[mt], false, B0, (short)0, acc0[mt], false, false);
                }
                if (doC1) {
                    const char* xrow1 = xrow0 + 16 * XPITCH * XESZ;
                    v16bf B1 = XB16
                        ? lds_ld16(reinterpret_cast<const bf16_t*>(xrow1),
                                   reinterpret_cast<const bf16_t*>(xrow1) + 8)
                        : cvt16(reinterpret_cast<const float*>(xrow1));
                    #pragma unroll
                    for (int mt = 0; mt < 4; ++mt)
                        acc1[mt] = __builtin_amdgcn_wmma_f32_16x16x32_bf16(
                            false, Am[mt], false, B1, (short)0, acc1[mt], false, false);
                }
            }

            __builtin_amdgcn_s_wait_tensorcnt(0);  // scalar; nonzero only in wave 0
            __syncthreads();
            buf ^= 1;
        }

        // fold superchunk into running max; mask out-of-range window columns
        const bool v0 = (cb + laneLo) < NWIN;
        const bool v1 = (cb + 16 + laneLo) < NWIN;
        #pragma unroll
        for (int mt = 0; mt < 4; ++mt)
            #pragma unroll
            for (int r = 0; r < 8; ++r) {
                vmax[mt][r] = fmaxf(vmax[mt][r], v0 ? acc0[mt][r] : -__builtin_inff());
                vmax[mt][r] = fmaxf(vmax[mt][r], v1 ? acc1[mt][r] : -__builtin_inff());
            }
    }

    // ---- reduce max over time: bias, cross-lane (16 cols), cross-wave ----
    #pragma unroll
    for (int mt = 0; mt < 4; ++mt) {
        #pragma unroll
        for (int r = 0; r < 8; ++r) {
            const int rowLocal = mt * 16 + (hiG ? 8 : 0) + r;   // C-layout row
            float v = vmax[mt][r] + bias[dBase + rowLocal];     // max commutes with +b
            #pragma unroll
            for (int off = 1; off < 16; off <<= 1)
                v = fmaxf(v, __shfl_xor(v, off, 32));
            if (laneLo == 0) red[wave * 64 + rowLocal] = v;
        }
    }
    __syncthreads();

    if (tid < 64) {
        float m = red[tid];
        #pragma unroll
        for (int w = 1; w < 8; ++w) m = fmaxf(m, red[w * 64 + tid]);
        m = fmaxf(m, 0.f);                                      // ReLU + max-init-0
        out[(size_t)bIdx * (3 * DEPTH) + outOff + dBase + tid] = m;
    }
}

template<int N, bool XB16>
static void launch_branch(const float* x, const bf16_t* xbf,
                          const float* W, const float* b,
                          float* out, int outOff, hipStream_t stream)
{
    constexpr int nK   = N * EMB;
    constexpr int Kpad = ((nK + 31) / 32) * 32;
    constexpr int KP   = Kpad + 8;
    constexpr int XPITCH = XB16 ? 40 : 36;
    constexpr int XESZ   = XB16 ? 2 : 4;
    constexpr size_t lds = (size_t)64 * KP * 2
                         + (size_t)2 * 256 * XPITCH * XESZ
                         + (size_t)8 * 64 * sizeof(float);
    hipFuncSetAttribute(reinterpret_cast<const void*>(&textcnn_wmma<N, XB16>),
                        hipFuncAttributeMaxDynamicSharedMemorySize, (int)lds);
    dim3 grid(DEPTH / 64, 64, 1);
    textcnn_wmma<N, XB16><<<grid, 256, lds, stream>>>(x, xbf, W, b, out, outOff);
}

extern "C" void kernel_launch(void* const* d_in, const int* in_sizes, int n_in,
                              void* d_out, int out_size, void* d_ws, size_t ws_size,
                              hipStream_t stream)
{
    const float* x  = (const float*)d_in[0];
    const float* W1 = (const float*)d_in[1];
    const float* W2 = (const float*)d_in[2];
    const float* W3 = (const float*)d_in[3];
    const float* b1 = (const float*)d_in[4];
    const float* b2 = (const float*)d_in[5];
    const float* b3 = (const float*)d_in[6];
    float* out = (float*)d_out;

    const size_t xElems   = (size_t)64 * SEQ * EMB;     // 7,564,800
    const size_t xbfBytes = xElems * sizeof(bf16_t);

    if (ws_size >= xbfBytes) {
        // pre-convert x to bf16 once; hot loops then TDM bf16 with no cvt
        bf16_t* xbf = (bf16_t*)d_ws;
        const int n4 = (int)(xElems / 4);
        cvt_x_bf16<<<(n4 + 255) / 256, 256, 0, stream>>>(x, xbf, n4);
        launch_branch<3, true>(x, xbf, W1, b1, out, 0,         stream);
        launch_branch<4, true>(x, xbf, W2, b2, out, DEPTH,     stream);
        launch_branch<5, true>(x, xbf, W3, b3, out, 2 * DEPTH, stream);
    } else {
        // fallback: TDM f32 tiles, convert at operand load
        launch_branch<3, false>(x, nullptr, W1, b1, out, 0,         stream);
        launch_branch<4, false>(x, nullptr, W2, b2, out, DEPTH,     stream);
        launch_branch<5, false>(x, nullptr, W3, b3, out, 2 * DEPTH, stream);
    }
}
